// SimpleGRU_57629871178454
// MI455X (gfx1250) — compile-verified, matrix operand order checked by reading, not run on hardware
//
#include <hip/hip_runtime.h>

// CDNA5 / gfx1250 WMMA GRU.
// One wave (32 lanes) owns a 16-batch tile. Per timestep:
//   D[q] = v_wmma_f32_16x16x32_f16(Whh_tile[q], h_f16, bias[q])   (6 WMMAs, K=32==H)
// Gate math is lane-local (D-layout == next step's B-layout); h never leaves registers.
// Activations use gfx1250's hardware v_tanh_f32 when available.

typedef __attribute__((ext_vector_type(16))) _Float16 v16h;
typedef __attribute__((ext_vector_type(8)))  float    v8f;

#define GRU_T 512
#define GRU_H 32

#if __has_builtin(__builtin_amdgcn_tanhf)
__device__ __forceinline__ float hw_tanh(float x) { return __builtin_amdgcn_tanhf(x); }
#else
__device__ __forceinline__ float hw_tanh(float x) {
  // tanh(x) = 2/(1+exp(-2x)) - 1 via v_exp_f32 + v_rcp_f32
  float e = __builtin_amdgcn_exp2f(-2.88538008177792681472f * x);
  float s = __builtin_amdgcn_rcpf(1.0f + e);
  return __builtin_fmaf(2.0f, s, -1.0f);
}
#endif

__device__ __forceinline__ float hw_sigmoid(float x) {
  // sigmoid(x) = 0.5 + 0.5*tanh(0.5*x): 1 transcendental, no divide
  return __builtin_fmaf(0.5f, hw_tanh(0.5f * x), 0.5f);
}

__global__ __launch_bounds__(32)
void gru_wmma_kernel(const float* __restrict__ x,
                     const float* __restrict__ W_ih,
                     const float* __restrict__ W_hh,
                     const float* __restrict__ b_ih,
                     const float* __restrict__ b_hh,
                     const float* __restrict__ fc_w,
                     const float* __restrict__ fc_b,
                     float* __restrict__ out) {
  const int lane  = threadIdx.x;          // 0..31, full wave (EXEC all-ones for WMMA)
  const int col   = lane & 15;            // batch column within tile / gate row for A
  const int kbase = (lane < 16) ? 0 : 8;  // K (hidden) sub-range this lane owns
  const int mofs  = kbase;                // M sub-range in C/D tiles
  const int b     = blockIdx.x * 16 + col;

  // ---- A operand: W_hh as six 16x32 f16 tiles (gate rows x hidden K) ----
  // 16-bit A layout: lane<16 -> K in {0..7,16..23}; lane>=16 -> K in {8..15,24..31};
  // element e maps to K = kbase + (e<8 ? e : e+8).
  v16h A[6];
#pragma unroll
  for (int q = 0; q < 6; ++q) {
    const int row = 16 * q + col;
#pragma unroll
    for (int e = 0; e < 16; ++e) {
      const int k = kbase + (e < 8 ? e : e + 8);
      A[q][e] = (_Float16)W_hh[row * GRU_H + k];
    }
  }

  // ---- C operand: biases pre-placed in f32 C/D layout ----
  // r,z tiles (q=0..3): b_ih+b_hh ; n tiles (q=4,5): b_hh only (PyTorch n-gate rule).
  v8f Cb[6];
#pragma unroll
  for (int q = 0; q < 6; ++q) {
#pragma unroll
    for (int v = 0; v < 8; ++v) {
      const int g = 16 * q + mofs + v;
      Cb[q][v] = (q < 4) ? (b_ih[g] + b_hh[g]) : b_hh[g];
    }
  }

  // Input-projection coefficients (I==1: xg = x*W_ih[g] (+ b_ih folded above for r,z))
  float wx[4][8];                     // r,z tiles, mirrors D layout
#pragma unroll
  for (int q = 0; q < 4; ++q)
#pragma unroll
    for (int v = 0; v < 8; ++v) wx[q][v] = W_ih[16 * q + mofs + v];

  float xnw[16], xnb[16];             // n-gate x-projection per hidden slot e
#pragma unroll
  for (int e = 0; e < 16; ++e) {
    const int j = kbase + (e < 8 ? e : e + 8);
    xnw[e] = W_ih[64 + j];
    xnb[e] = b_ih[64 + j];
  }

  // ---- recurrent state: h lives as f16 in B-operand layout, h0 = 0 ----
  v16h hB = {};

  const float* xrow = x + (size_t)b * GRU_T;  // I == 1; contiguous per-lane stream

  // software-pipelined x stream: load t0+4 while computing t0 (wrap keeps it in-bounds)
  float4 xc = *(const float4*)(xrow);
  for (int t0 = 0; t0 < GRU_T; t0 += 4) {
    const float4 xnx = *(const float4*)(xrow + ((t0 + 4) & (GRU_T - 1)));
    __builtin_prefetch(xrow + ((t0 + 64) & (GRU_T - 1)), 0, 0);  // global_prefetch_b8

#pragma unroll
    for (int s = 0; s < 4; ++s) {
      const float xb = (s == 0) ? xc.x : (s == 1) ? xc.y : (s == 2) ? xc.z : xc.w;

      v8f D[6];
#pragma unroll
      for (int q = 0; q < 6; ++q)
        D[q] = __builtin_amdgcn_wmma_f32_16x16x32_f16(
            /*neg_a=*/false, A[q], /*neg_b=*/false, hB,
            /*c_mod=*/(short)0, Cb[q], /*reuse_a=*/false, /*reuse_b=*/false);

      // add x * W_ih into r,z gate pre-activations (off the WMMA critical path)
#pragma unroll
      for (int q = 0; q < 4; ++q)
#pragma unroll
        for (int v = 0; v < 8; ++v)
          D[q][v] = __builtin_fmaf(xb, wx[q][v], D[q][v]);

      // lane-local GRU update; result written straight into next B operand
      v16h hN;
#pragma unroll
      for (int e = 0; e < 16; ++e) {
        const float rv = (e < 8) ? D[0][e] : D[1][e - 8];
        const float zv = (e < 8) ? D[2][e] : D[3][e - 8];
        const float hn = (e < 8) ? D[4][e] : D[5][e - 8];   // W_hh_n@h + b_hh_n
        const float r  = hw_sigmoid(rv);
        const float z  = hw_sigmoid(zv);
        const float xn = __builtin_fmaf(xb, xnw[e], xnb[e]);
        const float n  = hw_tanh(__builtin_fmaf(r, hn, xn));
        const float hp = (float)hB[e];
        const float h2 = __builtin_fmaf(z, hp - n, n);      // (1-z)*n + z*h
        hN[e] = (_Float16)h2;
      }
      hB = hN;
    }
    xc = xnx;
  }

  // ---- final FC: out[b, c] = h . fc_w[c] + fc_b[c], C = 2 ----
  float p0 = 0.f, p1 = 0.f;
#pragma unroll
  for (int e = 0; e < 16; ++e) {
    const int j = kbase + (e < 8 ? e : e + 8);
    const float hv = (float)hB[e];
    p0 = __builtin_fmaf(hv, fc_w[j], p0);
    p1 = __builtin_fmaf(hv, fc_w[GRU_H + j], p1);
  }
  p0 += __shfl_xor(p0, 16, 32);   // combine the two half-lane partial sums
  p1 += __shfl_xor(p1, 16, 32);
  if (lane < 16) {
    float2 o = make_float2(p0 + fc_b[0], p1 + fc_b[1]);
    *(float2*)(out + (size_t)b * 2) = o;
  }
}

extern "C" void kernel_launch(void* const* d_in, const int* in_sizes, int n_in,
                              void* d_out, int out_size, void* d_ws, size_t ws_size,
                              hipStream_t stream) {
  const float* x    = (const float*)d_in[0];
  const float* W_ih = (const float*)d_in[1];
  const float* W_hh = (const float*)d_in[2];
  const float* b_ih = (const float*)d_in[3];
  const float* b_hh = (const float*)d_in[4];
  const float* fc_w = (const float*)d_in[5];
  const float* fc_b = (const float*)d_in[6];
  float* out        = (float*)d_out;

  const int B = 4096;
  dim3 grid(B / 16), block(32);   // one full wave per 16-batch tile; 256 independent waves
  gru_wmma_kernel<<<grid, block, 0, stream>>>(x, W_ih, W_hh, b_ih, b_hh, fc_w, fc_b, out);
}
